// CrossDimensionalAttention_60550448939365
// MI455X (gfx1250) — matrix operand order repaired and checked by reading, not compiled, not run on hardware
//
#include <hip/hip_runtime.h>
#include <stdint.h>

#define Hdim 512
#define Bdim 4
#define Sdim 2048
#define EPS 1e-5f
#define HP  (Hdim + 8)   // padded LDS row stride (ushorts) -> conflict-free b128 A loads

typedef __attribute__((ext_vector_type(16))) __bf16 v16bf;
typedef __attribute__((ext_vector_type(8)))  float  v8f;

union Frag { v16bf bf; uint32_t u[8]; };

__device__ __forceinline__ unsigned short f2bf(float x) {
    uint32_t u = __builtin_bit_cast(uint32_t, x);
    u += 0x7FFFu + ((u >> 16) & 1u);           // round-to-nearest-even
    return (unsigned short)(u >> 16);
}
__device__ __forceinline__ float bf2f(unsigned short h) {
    uint32_t u = ((uint32_t)h) << 16;
    return __builtin_bit_cast(float, u);
}

// ---------------- Kernel 1: fp[b,k] = static[b,:] . Wf[k,:] + bf[k] ----------------
__global__ __launch_bounds__(256) void fp_kernel(const float* __restrict__ stat,
                                                 const float* __restrict__ Wf,
                                                 const float* __restrict__ bfv,
                                                 float* __restrict__ fp) {
    int idx = blockIdx.x * 256 + threadIdx.x;        // 0 .. B*H-1
    int b = idx >> 9, k = idx & (Hdim - 1);
    const float4* s4 = (const float4*)(stat + (size_t)b * Hdim);
    const float4* w4 = (const float4*)(Wf + (size_t)k * Hdim);
    float acc = bfv[k];
    #pragma unroll 4
    for (int i = 0; i < Hdim / 4; ++i) {
        float4 s = s4[i], w = w4[i];
        acc += s.x * w.x + s.y * w.y + s.z * w.z + s.w * w.w;
    }
    fp[idx] = acc;
}

// ---------------- Kernel 2: split Wo into bf16 hi/lo planes ----------------
__global__ __launch_bounds__(256) void split_kernel(const float* __restrict__ Wo,
                                                    unsigned short* __restrict__ hi,
                                                    unsigned short* __restrict__ lo) {
    int i = blockIdx.x * 256 + threadIdx.x;          // 0 .. H*H-1
    float x = Wo[i];
    unsigned short h = f2bf(x);
    hi[i] = h;
    lo[i] = f2bf(x - bf2f(h));
}

// ---------------- Kernel 3: fused  LN1 -> GEMM(WMMA bf16x3) -> LN2 ----------------
__global__ __launch_bounds__(256) void fused_kernel(const float* __restrict__ temporal,
                                                    const float* __restrict__ fp,
                                                    const unsigned short* __restrict__ WoHi,
                                                    const unsigned short* __restrict__ WoLo,
                                                    const float* __restrict__ bo,
                                                    const float* __restrict__ g1,
                                                    const float* __restrict__ b1,
                                                    const float* __restrict__ g2,
                                                    const float* __restrict__ b2,
                                                    float* __restrict__ out) {
    // aBuf[0]=y1 hi, aBuf[1]=y1 lo (bf16). Reused later as z (f32). 33,280 bytes.
    __shared__ unsigned short aBuf[2][16 * HP];

    const int tile = blockIdx.x;            // one 16-row tile of the 8192-row problem
    const int r0   = tile * 16;
    const int b    = r0 / Sdim;
    const int lane = threadIdx.x & 31;
    const int wave = threadIdx.x >> 5;      // 0..7

    // ---------- Phase 1: y1 = LN(temporal + fp[b]) ; store bf16 hi/lo to LDS ----------
    const float4* frow4 = (const float4*)(fp + (size_t)b * Hdim);
    for (int rr = 0; rr < 2; ++rr) {
        const int m = wave * 2 + rr;
        const float4* xrow4 = (const float4*)(temporal + (size_t)(r0 + m) * Hdim);
        float v[16];
        float sum = 0.f, sq = 0.f;
        #pragma unroll
        for (int i = 0; i < 4; ++i) {
            float4 x = xrow4[lane * 4 + i];
            float4 f = frow4[lane * 4 + i];
            float a0 = x.x + f.x, a1 = x.y + f.y, a2 = x.z + f.z, a3 = x.w + f.w;
            v[4 * i + 0] = a0; v[4 * i + 1] = a1; v[4 * i + 2] = a2; v[4 * i + 3] = a3;
            sum += a0 + a1 + a2 + a3;
            sq  += a0 * a0 + a1 * a1 + a2 * a2 + a3 * a3;
        }
        #pragma unroll
        for (int off = 16; off >= 1; off >>= 1) {
            sum += __shfl_xor(sum, off, 32);
            sq  += __shfl_xor(sq,  off, 32);
        }
        const float mean = sum * (1.0f / Hdim);
        const float var  = sq * (1.0f / Hdim) - mean * mean;
        const float rs   = rsqrtf(var + EPS);

        uint32_t* hiRow = (uint32_t*)&aBuf[0][m * HP];
        uint32_t* loRow = (uint32_t*)&aBuf[1][m * HP];
        #pragma unroll
        for (int i = 0; i < 4; ++i) {
            float4 g = ((const float4*)g1)[lane * 4 + i];
            float4 bb = ((const float4*)b1)[lane * 4 + i];
            float y0 = (v[4*i+0] - mean) * rs * g.x + bb.x;
            float y1 = (v[4*i+1] - mean) * rs * g.y + bb.y;
            float y2 = (v[4*i+2] - mean) * rs * g.z + bb.z;
            float y3 = (v[4*i+3] - mean) * rs * g.w + bb.w;
            unsigned short h0 = f2bf(y0), h1 = f2bf(y1), h2 = f2bf(y2), h3 = f2bf(y3);
            hiRow[lane * 8 + 2 * i + 0] = (uint32_t)h0 | ((uint32_t)h1 << 16);
            hiRow[lane * 8 + 2 * i + 1] = (uint32_t)h2 | ((uint32_t)h3 << 16);
            unsigned short l0 = f2bf(y0 - bf2f(h0)), l1 = f2bf(y1 - bf2f(h1));
            unsigned short l2 = f2bf(y2 - bf2f(h2)), l3 = f2bf(y3 - bf2f(h3));
            loRow[lane * 8 + 2 * i + 0] = (uint32_t)l0 | ((uint32_t)l1 << 16);
            loRow[lane * 8 + 2 * i + 1] = (uint32_t)l2 | ((uint32_t)l3 << 16);
        }
    }
    __syncthreads();

    // ---------- Phase 2: proj = y1 @ Wo^T  (bf16x3 split, f32 accumulate) ----------
    // A layout (16-bit 16x32): lanes 0-15 -> K {0..7, 16..23}; lanes 16-31 -> K {8..15, 24..31}
    // B layout (16-bit 32x16): lanes 0-15 -> K {0..7, 16..23}; lanes 16-31 -> K {8..15, 24..31}; N = lane%16
    const int am   = lane & 15;
    const int aOff = (lane < 16) ? 0 : 8;
    const int bn   = lane & 15;
    const int bOff = (lane < 16) ? 0 : 8;

    v8f c[4];
    #pragma unroll
    for (int t = 0; t < 4; ++t)
        #pragma unroll
        for (int r = 0; r < 8; ++r) c[t][r] = 0.0f;

    for (int k0 = 0; k0 < Hdim; k0 += 32) {
        Frag ah, al;
        *(uint4*)&ah.u[0] = *(const uint4*)&aBuf[0][am * HP + k0 + aOff];
        *(uint4*)&ah.u[4] = *(const uint4*)&aBuf[0][am * HP + k0 + aOff + 16];
        *(uint4*)&al.u[0] = *(const uint4*)&aBuf[1][am * HP + k0 + aOff];
        *(uint4*)&al.u[4] = *(const uint4*)&aBuf[1][am * HP + k0 + aOff + 16];
        #pragma unroll
        for (int t = 0; t < 4; ++t) {
            const int n = (wave + t * 8) * 16 + bn;
            const unsigned short* hrow = WoHi + (size_t)n * Hdim + k0 + bOff;
            const unsigned short* lrow = WoLo + (size_t)n * Hdim + k0 + bOff;
            Frag bh, bl;
            *(uint4*)&bh.u[0] = *(const uint4*)&hrow[0];
            *(uint4*)&bh.u[4] = *(const uint4*)&hrow[16];
            *(uint4*)&bl.u[0] = *(const uint4*)&lrow[0];
            *(uint4*)&bl.u[4] = *(const uint4*)&lrow[16];
            c[t] = __builtin_amdgcn_wmma_f32_16x16x32_bf16(false, al.bf, false, bh.bf,
                                                           (short)0, c[t], false, false);
            c[t] = __builtin_amdgcn_wmma_f32_16x16x32_bf16(false, ah.bf, false, bl.bf,
                                                           (short)0, c[t], false, false);
            c[t] = __builtin_amdgcn_wmma_f32_16x16x32_bf16(false, ah.bf, false, bh.bf,
                                                           (short)0, c[t], false, false);
        }
    }
    __syncthreads();

    // ---------- Phase 3: z = y1 + proj + bo   (y1 reconstructed = hi+lo) ----------
    const int mb = (lane < 16) ? 0 : 8;     // C/D layout: VGPR r -> M = r + (lane<16?0:8), N = lane%16
    float yv[4][8];
    #pragma unroll
    for (int t = 0; t < 4; ++t) {
        const int n = (wave + t * 8) * 16 + bn;
        #pragma unroll
        for (int r = 0; r < 8; ++r) {
            const int m = mb + r;
            yv[t][r] = bf2f(aBuf[0][m * HP + n]) + bf2f(aBuf[1][m * HP + n]);
        }
    }
    __syncthreads();
    float* z = (float*)&aBuf[0][0];         // 16*512 f32 = 32 KB, fits in aBuf
    #pragma unroll
    for (int t = 0; t < 4; ++t) {
        const int n = (wave + t * 8) * 16 + bn;
        const float bon = bo[n];
        #pragma unroll
        for (int r = 0; r < 8; ++r)
            z[(mb + r) * Hdim + n] = yv[t][r] + c[t][r] + bon;
    }
    __syncthreads();

    // ---------- Phase 4: out = LN(z) ----------
    for (int rr = 0; rr < 2; ++rr) {
        const int m = wave * 2 + rr;
        float v[16];
        float sum = 0.f, sq = 0.f;
        #pragma unroll
        for (int i = 0; i < 4; ++i) {
            float4 x = ((const float4*)&z[m * Hdim])[lane * 4 + i];
            v[4*i+0] = x.x; v[4*i+1] = x.y; v[4*i+2] = x.z; v[4*i+3] = x.w;
            sum += x.x + x.y + x.z + x.w;
            sq  += x.x * x.x + x.y * x.y + x.z * x.z + x.w * x.w;
        }
        #pragma unroll
        for (int off = 16; off >= 1; off >>= 1) {
            sum += __shfl_xor(sum, off, 32);
            sq  += __shfl_xor(sq,  off, 32);
        }
        const float mean = sum * (1.0f / Hdim);
        const float var  = sq * (1.0f / Hdim) - mean * mean;
        const float rs   = rsqrtf(var + EPS);
        float4* orow = (float4*)(out + (size_t)(r0 + m) * Hdim);
        #pragma unroll
        for (int i = 0; i < 4; ++i) {
            float4 g = ((const float4*)g2)[lane * 4 + i];
            float4 bb = ((const float4*)b2)[lane * 4 + i];
            float4 o;
            o.x = (v[4*i+0] - mean) * rs * g.x + bb.x;
            o.y = (v[4*i+1] - mean) * rs * g.y + bb.y;
            o.z = (v[4*i+2] - mean) * rs * g.z + bb.z;
            o.w = (v[4*i+3] - mean) * rs * g.w + bb.w;
            orow[lane * 4 + i] = o;
        }
    }
}

extern "C" void kernel_launch(void* const* d_in, const int* in_sizes, int n_in,
                              void* d_out, int out_size, void* d_ws, size_t ws_size,
                              hipStream_t stream) {
    (void)in_sizes; (void)n_in; (void)out_size; (void)ws_size;
    const float* temporal = (const float*)d_in[0];
    const float* stat     = (const float*)d_in[1];
    // d_in[2]=Wt, d_in[3]=bt are dead: softmax over a constant axis is uniform,
    // so attended == fp broadcast and tp never affects the output.
    const float* Wf  = (const float*)d_in[4];
    const float* bfv = (const float*)d_in[5];
    const float* Wo  = (const float*)d_in[6];
    const float* bo  = (const float*)d_in[7];
    const float* g1  = (const float*)d_in[8];
    const float* b1  = (const float*)d_in[9];
    const float* g2  = (const float*)d_in[10];
    const float* b2  = (const float*)d_in[11];
    float* out = (float*)d_out;

    char* ws = (char*)d_ws;
    float* fp = (float*)ws;                                   // B*H f32 = 8 KB
    unsigned short* WoHi = (unsigned short*)(ws + 16384);     // 512 KB
    unsigned short* WoLo = WoHi + (size_t)Hdim * Hdim;        // 512 KB

    fp_kernel<<<(Bdim * Hdim) / 256, 256, 0, stream>>>(stat, Wf, bfv, fp);
    split_kernel<<<(Hdim * Hdim) / 256, 256, 0, stream>>>(Wo, WoHi, WoLo);
    fused_kernel<<<(Bdim * Sdim) / 16, 256, 0, stream>>>(temporal, fp, WoHi, WoLo,
                                                         bo, g1, b1, g2, b2, out);
}
